// Splat2DModule_13460427505822
// MI455X (gfx1250) — compile-verified, matrix operand order checked by reading, not run on hardware
//
#include <hip/hip_runtime.h>

// CDNA5 (gfx1250) wave32 WMMA types
typedef __attribute__((ext_vector_type(16))) _Float16 v16h;
typedef __attribute__((ext_vector_type(8)))  float    v8f;
typedef __attribute__((ext_vector_type(2)))  float    v2f;

#define RES      512
#define NBLOBS   2048
#define KTILE    32                 // blobs per f16 WMMA (K of 16x16x32)
#define NITER    (NBLOBS / KTILE)   // 64
#define THREADS  256                // 8 wave32 per block
#define PIX_PER_WAVE  32            // two 16-pixel WMMA tiles per wave
#define PIX_PER_BLOCK (8 * PIX_PER_WAVE)   // 256

// LDS layout (80 KB):
//   [0,     65536) : float coef[2048][8] = (A,B,C,D,E,F,0,0)   (scaled)
//   [65536, 81920) : _Float16 col[2048][4] = (r, g, b, 0)
// q(px,py) = A*px^2 + B*px*py + C*py^2 + D*px + E*py + F, pre-scaled by
// -0.5*log2(e) so weight = exp2(q) directly.
#define SMEM_BYTES (81920)

__global__ __launch_bounds__(THREADS)
void splat2d_wmma_kernel(const float* __restrict__ blobs,
                         float* __restrict__ out) {
  extern __shared__ char smem_raw[];
  float4*   scoef4 = reinterpret_cast<float4*>(smem_raw);
  v2f*      scoef2 = reinterpret_cast<v2f*>(smem_raw);
  _Float16* scol   = reinterpret_cast<_Float16*>(smem_raw + 65536);

  // ---- Stage per-blob quadratic coefficients + f16 colors into LDS ----
  // blobs row: [posx, posy, sx, sy, rot, r, g, b]
  const float kExpScale = -0.72134752044448170368f; // -0.5 * log2(e)
  for (int i = 0; i < NBLOBS / THREADS; ++i) {
    const int j = threadIdx.x + i * THREADS;
    const float4* bp = reinterpret_cast<const float4*>(blobs + j * 8);
    const float4 b0 = bp[0];              // posx, posy, sx, sy
    const float4 b1 = bp[1];              // rot, r, g, b
    float sn, cs;
    __sincosf(b1.x, &sn, &cs);
    const float isx2 = __frcp_rn(b0.z * b0.z);
    const float isy2 = __frcp_rn(b0.w * b0.w);
    const float m11 = (cs * cs * isx2 + sn * sn * isy2) * kExpScale;
    const float m22 = (sn * sn * isx2 + cs * cs * isy2) * kExpScale;
    const float m12 = 2.0f * cs * sn * (isx2 - isy2) * kExpScale;
    const float X = b0.x, Y = b0.y;
    const float cA = m11;
    const float cB = m12;
    const float cC = m22;
    const float cD = -(2.0f * m11 * X + m12 * Y);
    const float cE = -(2.0f * m22 * Y + m12 * X);
    const float cF = m11 * X * X + m12 * X * Y + m22 * Y * Y;
    scoef4[j * 2 + 0] = make_float4(cA, cB, cC, cD);
    scoef4[j * 2 + 1] = make_float4(cE, cF, 0.0f, 0.0f);
    scol[j * 4 + 0] = (_Float16)b1.y;
    scol[j * 4 + 1] = (_Float16)b1.z;
    scol[j * 4 + 2] = (_Float16)b1.w;
    scol[j * 4 + 3] = (_Float16)0.0f;   // read by B-columns n >= 3
  }
  __syncthreads();

  const int lane = threadIdx.x & 31;
  const int wave = threadIdx.x >> 5;
  const int h16  = lane >> 4;           // which 16-lane half (h)
  const int n    = lane & 15;           // matrix row/column index for this lane
  const int h8   = h16 * 8;
  const int cidx = (n < 3) ? n : 3;     // branchless color index (slot 3 == 0)

  // Each wave owns 32 consecutive pixels: two 16-row tiles (A-tile, B-tile).
  const int base0 = blockIdx.x * PIX_PER_BLOCK + wave * PIX_PER_WAVE;
  const int pixA = base0 + n;
  const int pixB = base0 + 16 + n;
  const float pxA = ((pixA & (RES - 1)) + 0.5f) * (1.0f / RES);
  const float pyA = ((pixA >> 9)        + 0.5f) * (1.0f / RES);
  const float pxB = ((pixB & (RES - 1)) + 0.5f) * (1.0f / RES);
  const float pyB = ((pixB >> 9)        + 0.5f) * (1.0f / RES);

  // Loop-invariant monomial B-matrices for the f32 q-WMMA (4x16, 2 VGPRs).
  // f32 B layout: lanes 0-15 supply K=0 (vgpr0), K=1 (vgpr1);
  //               lanes 16-31 supply K=2 (vgpr0), K=3 (vgpr1).
  // chunk0 terms: K0=px^2, K1=px*py, K2=py^2, K3=px
  // chunk1 terms: K0=py,   K1=1,     K2=0,    K3=0
  const bool hi = (h16 != 0);
  v2f bm0A, bm1A, bm0B, bm1B;
  bm0A.x = hi ? (pyA * pyA) : (pxA * pxA);
  bm0A.y = hi ? pxA         : (pxA * pyA);
  bm1A.x = hi ? 0.0f        : pyA;
  bm1A.y = hi ? 0.0f        : 1.0f;
  bm0B.x = hi ? (pyB * pyB) : (pxB * pxB);
  bm0B.y = hi ? pxB         : (pxB * pyB);
  bm1B.x = hi ? 0.0f        : pyB;
  bm1B.y = hi ? 0.0f        : 1.0f;

  const v8f vzero = {0.f, 0.f, 0.f, 0.f, 0.f, 0.f, 0.f, 0.f};
  v8f acc0 = vzero;
  v8f acc1 = vzero;

  for (int it = 0; it < NITER; ++it) {
    const int blob0 = it * KTILE;

    // f32 A layout (16x4): lane supplies blob row M = lane&15; this half's
    // term pair is (2h, 2h+1) -> coefficient pair index h (chunk0) / 2+h.
    const int kg0 = blob0 + n;        // blob for group 0 (K = 0..15)
    const int kg1 = blob0 + 16 + n;   // blob for group 1 (K = 16..31)
    const v2f ac0_g0 = scoef2[kg0 * 4 + h16];
    const v2f ac1_g0 = scoef2[kg0 * 4 + 2 + h16];
    const v2f ac0_g1 = scoef2[kg1 * 4 + h16];
    const v2f ac1_g1 = scoef2[kg1 * 4 + 2 + h16];

    // q = Coef x Mono via chained f32 WMMAs (K=4 each, 6 terms + 2 pad).
    v8f qA0 = __builtin_amdgcn_wmma_f32_16x16x4_f32(
        false, ac0_g0, false, bm0A, (short)0, vzero, false, false);
    qA0 = __builtin_amdgcn_wmma_f32_16x16x4_f32(
        false, ac1_g0, false, bm1A, (short)0, qA0, false, false);
    v8f qA1 = __builtin_amdgcn_wmma_f32_16x16x4_f32(
        false, ac0_g1, false, bm0A, (short)0, vzero, false, false);
    qA1 = __builtin_amdgcn_wmma_f32_16x16x4_f32(
        false, ac1_g1, false, bm1A, (short)0, qA1, false, false);
    v8f qB0 = __builtin_amdgcn_wmma_f32_16x16x4_f32(
        false, ac0_g0, false, bm0B, (short)0, vzero, false, false);
    qB0 = __builtin_amdgcn_wmma_f32_16x16x4_f32(
        false, ac1_g0, false, bm1B, (short)0, qB0, false, false);
    v8f qB1 = __builtin_amdgcn_wmma_f32_16x16x4_f32(
        false, ac0_g1, false, bm0B, (short)0, vzero, false, false);
    qB1 = __builtin_amdgcn_wmma_f32_16x16x4_f32(
        false, ac1_g1, false, bm1B, (short)0, qB1, false, false);

    // q-WMMA D (lane = pixel col, vgpr r = blob row r+8h) lines up exactly
    // with the f16 A-matrix slots: slot s < 8 -> group0 row s (K = h8+s),
    // slot s >= 8 -> group1 row s-8 (K = 16+h8+s-8). No cross-lane moves.
    v16h a0, a1, b;
    #pragma unroll
    for (int s = 0; s < 8; ++s) {
      a0[s]     = (_Float16)__builtin_amdgcn_exp2f(qA0[s]);
      a0[s + 8] = (_Float16)__builtin_amdgcn_exp2f(qA1[s]);
      a1[s]     = (_Float16)__builtin_amdgcn_exp2f(qB0[s]);
      a1[s + 8] = (_Float16)__builtin_amdgcn_exp2f(qB1[s]);
    }

    // Branchless f16 color B tile (shared by both pixel tiles).
    #pragma unroll
    for (int s = 0; s < 16; ++s) {
      const int koff = (s < 8) ? (h8 + s) : (h8 + s + 8);
      b[s] = scol[(blob0 + koff) * 4 + cidx];
    }

    acc0 = __builtin_amdgcn_wmma_f32_16x16x32_f16(
        false, a0, false, b, (short)0, acc0, false, false);
    acc1 = __builtin_amdgcn_wmma_f32_16x16x32_f16(
        false, a1, false, b, (short)0, acc1, false, false);
  }

  // ---- Store D: lane holds column n; VGPR r holds row M = r + 8*h16 ----
  if (n < 3) {
    #pragma unroll
    for (int r = 0; r < 8; ++r) {
      out[(base0 + r + h8) * 3 + n]      = acc0[r];
      out[(base0 + 16 + r + h8) * 3 + n] = acc1[r];
    }
  }
}

extern "C" void kernel_launch(void* const* d_in, const int* in_sizes, int n_in,
                              void* d_out, int out_size, void* d_ws, size_t ws_size,
                              hipStream_t stream) {
  (void)in_sizes; (void)n_in; (void)out_size; (void)d_ws; (void)ws_size;
  const float* blobs = (const float*)d_in[0];   // 2048 x 8 f32
  float* out = (float*)d_out;                   // 512*512*3 f32

  dim3 grid((RES * RES) / PIX_PER_BLOCK);       // 1024 blocks
  dim3 block(THREADS);                          // 8 wave32
  hipLaunchKernelGGL(splat2d_wmma_kernel, grid, block, SMEM_BYTES, stream,
                     blobs, out);
}
